// VectorQuantizer_60370060313181
// MI455X (gfx1250) — matrix operand order repaired and checked by reading, not compiled, not run on hardware
//
#include <hip/hip_runtime.h>
#include <hip/hip_bf16.h>

// ---- types -----------------------------------------------------------------
typedef __attribute__((ext_vector_type(16))) __bf16 v16bf;
typedef __attribute__((ext_vector_type(8)))  float  v8f;

struct alignas(16) U4 { unsigned x, y, z, w; };
struct alignas(16) F4 { float x, y, z, w; };

union ABFrag {
    v16bf v;
    unsigned short s[16];
    U4 q[2];
};

#define NUM_CODES 1024
#define DIM       64
#define N_ROWS    32768   // 32*32*32

__device__ __forceinline__ unsigned short f2bf(float f) {
    unsigned u = __builtin_bit_cast(unsigned, f);
    u += 0x7fffu + ((u >> 16) & 1u);        // round-to-nearest-even
    return (unsigned short)(u >> 16);
}

// ---- kernel 1: codebook -> bf16, ||e||^2, zero accumulators ---------------
__global__ void __launch_bounds__(256) vq_prep(
    const float* __restrict__ emb, float* __restrict__ enorm,
    unsigned short* __restrict__ Ebf, unsigned* __restrict__ hist,
    float* __restrict__ commit) {
    int t = blockIdx.x * 256 + threadIdx.x;       // grid = 4*256 = 1024
    if (t >= NUM_CODES) return;
    float s = 0.0f;
    #pragma unroll 8
    for (int d = 0; d < DIM; ++d) {
        float v = emb[t * DIM + d];
        s += v * v;
        Ebf[t * DIM + d] = f2bf(v);
    }
    enorm[t] = s;
    hist[t] = 0u;
    if (t == 0) *commit = 0.0f;
}

// ---- kernel 2: WMMA distance sweep + argmin + gather ----------------------
__global__ void __launch_bounds__(256) vq_main(
    const float* __restrict__ latents, const float* __restrict__ emb,
    const unsigned short* __restrict__ Ebf, const float* __restrict__ enorm,
    unsigned* __restrict__ hist, float* __restrict__ commit,
    float* __restrict__ outq) {
    extern __shared__ char smem[];
    U4*    sEq = (U4*)smem;                  // 8192 * 16B = 128KB bf16 codebook
    float* sEn = (float*)(smem + 131072);    // 1024 floats = ||e||^2

    // ---- async stage: whole bf16 codebook + norms into LDS (ASYNCcnt) ----
    const U4* gE = (const U4*)Ebf;
    for (int i = threadIdx.x; i < 8192; i += 256) {
        unsigned lds = (unsigned)(unsigned long long)(sEq + i);
        asm volatile("global_load_async_to_lds_b128 %0, %1, off"
                     :: "v"(lds), "v"(gE + i) : "memory");
    }
    for (int i = threadIdx.x; i < 1024; i += 256) {
        unsigned lds = (unsigned)(unsigned long long)(sEn + i);
        asm volatile("global_load_async_to_lds_b32 %0, %1, off"
                     :: "v"(lds), "v"(enorm + i) : "memory");
    }
    asm volatile("s_wait_asynccnt 0x0" ::: "memory");
    __syncthreads();

    const int lane = threadIdx.x & 31;
    const int wave = threadIdx.x >> 5;
    const int m    = lane & 15;      // N / M sub-index within half-wave
    const int h    = lane >> 4;      // half select
    const int row_base = blockIdx.x * 256 + wave * 32;   // 2 tiles of 16 rows

    // Build A fragments: 2 tiles x (16 rows x 64 cols) fp32 -> bf16 frags.
    // ISA A layout (16-bit, 16x32): lane half h holds K in [8h,8h+8) U [16+8h,16+8h+8)
    ABFrag a[2][2];
    float xn2 = 0.0f;                         // per-lane sum of x^2 (exact coverage)
    const F4* Xv = (const F4*)latents;        // 16 float4 per row
    #pragma unroll
    for (int t = 0; t < 2; ++t) {
        long rb = (long)(row_base + t * 16 + m) * 16;
        #pragma unroll
        for (int kc = 0; kc < 2; ++kc) {
            F4 p[4];
            p[0] = Xv[rb + kc * 8 + 2 * h];
            p[1] = Xv[rb + kc * 8 + 2 * h + 1];
            p[2] = Xv[rb + kc * 8 + 4 + 2 * h];
            p[3] = Xv[rb + kc * 8 + 4 + 2 * h + 1];
            const float* e = (const float*)p;
            #pragma unroll
            for (int i = 0; i < 16; ++i) {
                a[t][kc].s[i] = f2bf(e[i]);
                xn2 += e[i] * e[i];
            }
        }
    }

    // Sweep all 64 code tiles; track argmin of (||e||^2 - 2 x.e) per row.
    float bestv[2][8];
    int   besti[2][8];
    #pragma unroll
    for (int t = 0; t < 2; ++t)
        #pragma unroll
        for (int r = 0; r < 8; ++r) { bestv[t][r] = 3.0e38f; besti[t][r] = 0; }

    const U4* sEq4 = sEq;
    // B layout (16-bit, 32x16): lane half h holds K in [8h,8h+8) U [16+8h,16+8h+8)
    // Prologue: fragments for code tile 0.
    ABFrag b0, b1;
    {
        int bi = m * 8 + h;                   // U4 index (8 shorts per U4)
        b0.q[0] = sEq4[bi];     b0.q[1] = sEq4[bi + 2];   // K 0..31
        b1.q[0] = sEq4[bi + 4]; b1.q[1] = sEq4[bi + 6];   // K 32..63
    }
    float en = sEn[m];

    for (int nt = 0; nt < 64; ++nt) {
        v8f acc0 = {0.f, 0.f, 0.f, 0.f, 0.f, 0.f, 0.f, 0.f};
        v8f acc1 = {0.f, 0.f, 0.f, 0.f, 0.f, 0.f, 0.f, 0.f};
        acc0 = __builtin_amdgcn_wmma_f32_16x16x32_bf16(
            false, a[0][0].v, false, b0.v, (short)0, acc0, false, false);
        acc0 = __builtin_amdgcn_wmma_f32_16x16x32_bf16(
            false, a[0][1].v, false, b1.v, (short)0, acc0, false, false);
        acc1 = __builtin_amdgcn_wmma_f32_16x16x32_bf16(
            false, a[1][0].v, false, b0.v, (short)0, acc1, false, false);
        acc1 = __builtin_amdgcn_wmma_f32_16x16x32_bf16(
            false, a[1][1].v, false, b1.v, (short)0, acc1, false, false);

        // software-pipelined prefetch of next code tile (wraps in-bounds)
        int cn = ((nt + 1) * 16 + m) & 1023;
        int bin = cn * 8 + h;
        ABFrag nb0, nb1;
        nb0.q[0] = sEq4[bin];     nb0.q[1] = sEq4[bin + 2];
        nb1.q[0] = sEq4[bin + 4]; nb1.q[1] = sEq4[bin + 6];
        float nen = sEn[cn];

        int c0 = nt * 16 + m;                 // this lane's candidate code id
        #pragma unroll
        for (int r = 0; r < 8; ++r) {         // row M = r + 8h, col N = m
            float v0 = en - 2.0f * acc0[r];
            float v1 = en - 2.0f * acc1[r];
            if (v0 < bestv[0][r]) { bestv[0][r] = v0; besti[0][r] = c0; }
            if (v1 < bestv[1][r]) { bestv[1][r] = v1; besti[1][r] = c0; }
        }
        b0 = nb0; b1 = nb1; en = nen;
    }

    // argmin across the 16 lanes of each half (ties -> lowest index)
    #pragma unroll
    for (int mask = 1; mask <= 8; mask <<= 1) {
        #pragma unroll
        for (int t = 0; t < 2; ++t)
            #pragma unroll
            for (int r = 0; r < 8; ++r) {
                float ov = __shfl_xor(bestv[t][r], mask, 32);
                int   oi = __shfl_xor(besti[t][r], mask, 32);
                if (ov < bestv[t][r] || (ov == bestv[t][r] && oi < besti[t][r])) {
                    bestv[t][r] = ov; besti[t][r] = oi;
                }
            }
    }

    // gather fp32 codebook rows -> output (coalesced: 16 lanes x float4)
    const F4* Ev = (const F4*)emb;
    F4* Ov = (F4*)outq;
    #pragma unroll
    for (int t = 0; t < 2; ++t)
        #pragma unroll
        for (int r = 0; r < 8; ++r) {
            int row = row_base + t * 16 + r + 8 * h;
            int idx = besti[t][r];
            Ov[(long)row * 16 + m] = Ev[(long)idx * 16 + m];
        }

    // histogram + commitment:  sum_rows ||x-q||^2 = sum x^2 + sum min(e2-2xe)
    float tsum = xn2;
    if (m == 0) {
        float s = 0.0f;
        #pragma unroll
        for (int t = 0; t < 2; ++t)
            #pragma unroll
            for (int r = 0; r < 8; ++r) {
                atomicAdd(&hist[besti[t][r]], 1u);
                s += bestv[t][r];
            }
        tsum += s;
    }
    #pragma unroll
    for (int mask = 16; mask >= 1; mask >>= 1) tsum += __shfl_xor(tsum, mask, 32);
    if (lane == 0) atomicAdd(commit, tsum);
}

// ---- kernel 3: perplexity + commitment scalars ----------------------------
__global__ void __launch_bounds__(1024) vq_finalize(
    const unsigned* __restrict__ hist, const float* __restrict__ commit,
    float* __restrict__ out) {
    __shared__ float red[32];
    int t = threadIdx.x;                       // 1024 threads, one per bin
    float p = (float)hist[t] * (1.0f / (float)N_ROWS);
    float term = p * logf(p + 1e-10f);
    #pragma unroll
    for (int mask = 16; mask >= 1; mask >>= 1) term += __shfl_xor(term, mask, 32);
    if ((t & 31) == 0) red[t >> 5] = term;
    __syncthreads();
    if (t < 32) {
        float v = red[t];
        #pragma unroll
        for (int mask = 16; mask >= 1; mask >>= 1) v += __shfl_xor(v, mask, 32);
        if (t == 0) {
            out[(long)N_ROWS * DIM]     = 0.25f * (*commit) / ((float)N_ROWS * (float)DIM);
            out[(long)N_ROWS * DIM + 1] = expf(-v);
        }
    }
}

// ---- launch ---------------------------------------------------------------
extern "C" void kernel_launch(void* const* d_in, const int* in_sizes, int n_in,
                              void* d_out, int out_size, void* d_ws, size_t ws_size,
                              hipStream_t stream) {
    const float* latents = (const float*)d_in[0];   // (32,32,32,64) fp32
    const float* emb     = (const float*)d_in[1];   // (1024,64) fp32
    float* out = (float*)d_out;

    char* ws = (char*)d_ws;
    float*          enorm  = (float*)ws;                       // 4 KB
    unsigned*       hist   = (unsigned*)(ws + 4096);           // 4 KB
    float*          commit = (float*)(ws + 8192);              // 4 B
    unsigned short* Ebf    = (unsigned short*)(ws + 8448);     // 128 KB bf16 codebook

    vq_prep<<<4, 256, 0, stream>>>(emb, enorm, Ebf, hist, commit);
    vq_main<<<128, 256, 135168, stream>>>(latents, emb, Ebf, enorm, hist, commit, out);
    vq_finalize<<<1, 1024, 0, stream>>>(hist, commit, out);
}
